// ArchitectureGNN_86105504350742
// MI455X (gfx1250) — matrix operand order repaired and checked by reading, not compiled, not run on hardware
//
#include <hip/hip_runtime.h>
#include <hip/hip_bf16.h>
#include <math.h>

// ---------------------------------------------------------------------------
// GAT (heads=4, hid=128, concat=False) x3  +  mean-pool  +  MLP, for gfx1250.
// Dense GEMMs use v_wmma_f32_16x16x32_bf16 with 16x64 per-wave register
// blocking (A-frag reused across 4 WMMAs). Edge softmax/aggregation use
// L2-resident float atomics (all hot buffers fit the 192MB L2).
// ---------------------------------------------------------------------------

#define HEADS 4
#define HID 128
#define NOUT 512          // HEADS*HID
#define KP 128            // padded K for all GEMMs (layer1: 12 -> zero pad)
#define NEG_SLOPE 0.2f

typedef __attribute__((ext_vector_type(16))) __bf16        v16bf;
typedef __attribute__((ext_vector_type(8)))  float         v8f;
typedef __attribute__((ext_vector_type(4)))  unsigned int  v4u;

union Frag16 {
    v16bf bf;
    struct { v4u lo; v4u hi; } u;
};

__device__ __forceinline__ unsigned short f2bf(float f) {
    unsigned int u = __float_as_uint(f);
    unsigned int r = u + 0x7FFFu + ((u >> 16) & 1u);   // round-to-nearest-even
    return (unsigned short)(r >> 16);
}

__device__ __forceinline__ float lrelu(float v) {
    return v > 0.0f ? v : NEG_SLOPE * v;
}

// ---- fp32 x [N,inDim] -> bf16 A [N,KP] zero-padded -------------------------
__global__ void k_cast_x(const float* __restrict__ x, unsigned short* __restrict__ A, int inDim) {
    int n = blockIdx.x, c = threadIdx.x;
    A[(size_t)n * KP + c] = (c < inDim) ? f2bf(x[(size_t)n * inDim + c]) : (unsigned short)0;
}

// ---- W [Kw,512] fp32 -> WT bf16 [512,KP] (transposed, zero-padded) ---------
__global__ void k_cast_wt(const float* __restrict__ W, unsigned short* __restrict__ WT, int Kw) {
    int col = blockIdx.x, k = threadIdx.x;               // 512 blocks x 128 thr
    WT[(size_t)col * KP + k] = (k < Kw) ? f2bf(W[(size_t)k * NOUT + col]) : (unsigned short)0;
}

// ---- H[M,512] = A[M,KP](bf16) @ WT[512,KP]^T via v_wmma_f32_16x16x32_bf16 --
// grid (M/16), block 256 (8 waves). Wave w computes the 16x64 strip
// [mbase:mbase+16, w*64:(w+1)*64]: A-frag loaded once per K-step, reused
// across 4 WMMAs (4x less A traffic, 16 WMMAs per kernel).
__global__ void k_gemm_bf16(const unsigned short* __restrict__ A,
                            const unsigned short* __restrict__ WT,
                            float* __restrict__ H) {
    const int tid  = threadIdx.x;
    const int wave = tid >> 5;
    const int lane = tid & 31;
    const int ll   = lane & 15;
    const int lh   = lane >> 4;
    const int mbase = blockIdx.x * 16;
    const int nbase = wave * 64;

    // A-frag: lane row = mbase+ll ; contiguous 16-K strip at lh*16
    const unsigned short* arow = A  + (size_t)(mbase + ll) * KP + (lh << 4);
    // B-frag t: lane col = nbase+16t+ll ; contiguous 16-K strip at lh*16
    const unsigned short* bcol = WT + (size_t)(nbase + ll) * KP + (lh << 4);

    v8f acc0 = {}, acc1 = {}, acc2 = {}, acc3 = {};
#pragma unroll
    for (int kk = 0; kk < KP; kk += 32) {
        __builtin_prefetch(arow + kk + 32, 0, 3);
        __builtin_prefetch(bcol + kk + 32, 0, 3);
        Frag16 a, b0, b1, b2, b3;
        a.u.lo  = *(const v4u*)(arow + kk);
        a.u.hi  = *(const v4u*)(arow + kk + 8);
        b0.u.lo = *(const v4u*)(bcol + kk);
        b0.u.hi = *(const v4u*)(bcol + kk + 8);
        b1.u.lo = *(const v4u*)(bcol + 16 * KP + kk);
        b1.u.hi = *(const v4u*)(bcol + 16 * KP + kk + 8);
        b2.u.lo = *(const v4u*)(bcol + 32 * KP + kk);
        b2.u.hi = *(const v4u*)(bcol + 32 * KP + kk + 8);
        b3.u.lo = *(const v4u*)(bcol + 48 * KP + kk);
        b3.u.hi = *(const v4u*)(bcol + 48 * KP + kk + 8);
        acc0 = __builtin_amdgcn_wmma_f32_16x16x32_bf16(
            false, a.bf, false, b0.bf, (short)0, acc0, false, false);
        acc1 = __builtin_amdgcn_wmma_f32_16x16x32_bf16(
            false, a.bf, false, b1.bf, (short)0, acc1, false, false);
        acc2 = __builtin_amdgcn_wmma_f32_16x16x32_bf16(
            false, a.bf, false, b2.bf, (short)0, acc2, false, false);
        acc3 = __builtin_amdgcn_wmma_f32_16x16x32_bf16(
            false, a.bf, false, b3.bf, (short)0, acc3, false, false);
    }

    // C/D layout: VGPR r -> row r + 8*lh, col = ll
    float* out = H + (size_t)(mbase + (lh << 3)) * NOUT + nbase + ll;
#pragma unroll
    for (int r = 0; r < 8; ++r) {
        out[(size_t)r * NOUT]      = acc0[r];
        out[(size_t)r * NOUT + 16] = acc1[r];
        out[(size_t)r * NOUT + 32] = acc2[r];
        out[(size_t)r * NOUT + 48] = acc3[r];
    }
}

// ---- attention scores: ES[n,h] = <H[n,h,:], a_src[h,:]> (ED likewise) ------
// one block per node, wave w = head w, lane handles 4 channels
__global__ void k_scores(const float* __restrict__ H, const float* __restrict__ asrc,
                         const float* __restrict__ adst,
                         float* __restrict__ ES, float* __restrict__ ED) {
    int n = blockIdx.x, t = threadIdx.x;
    int w = t >> 5, l = t & 31;
    const float4 hv = *(const float4*)(H + (size_t)n * NOUT + w * HID + l * 4);
    const float4 av = *(const float4*)(asrc + w * HID + l * 4);
    const float4 dv = *(const float4*)(adst + w * HID + l * 4);
    float es = hv.x * av.x + hv.y * av.y + hv.z * av.z + hv.w * av.w;
    float ed = hv.x * dv.x + hv.y * dv.y + hv.z * dv.z + hv.w * dv.w;
#pragma unroll
    for (int off = 16; off > 0; off >>= 1) {
        es += __shfl_down(es, off, 32);
        ed += __shfl_down(ed, off, 32);
    }
    if (l == 0) { ES[n * HEADS + w] = es; ED[n * HEADS + w] = ed; }
}

// ---- per-layer init: AGG=0, MX=-inf, DEN=0 ---------------------------------
__global__ void k_init(float* __restrict__ AGG, float* __restrict__ MX,
                       float* __restrict__ DEN, long total, int small) {
    long i = (long)blockIdx.x * blockDim.x + threadIdx.x;
    if (i < total) AGG[i] = 0.0f;
    if (i < small) { MX[i] = -1e30f; DEN[i] = 0.0f; }
}

// ---- segment max over incoming edges ---------------------------------------
__global__ void k_edge_max(const int* __restrict__ src, const int* __restrict__ dst,
                           const float* __restrict__ ES, const float* __restrict__ ED,
                           float* __restrict__ MX, int E, int ET) {
    int e = blockIdx.x * blockDim.x + threadIdx.x;
    if (e >= ET) return;
    int s = (e < E) ? src[e] : (e - E);
    int d = (e < E) ? dst[e] : (e - E);
#pragma unroll
    for (int h = 0; h < HEADS; ++h) {
        float v = lrelu(ES[s * HEADS + h] + ED[d * HEADS + h]);
        atomicMax(&MX[d * HEADS + h], v);
    }
}

// ---- exp(e - m[dst]) and segment sum ---------------------------------------
__global__ void k_edge_expsum(const int* __restrict__ src, const int* __restrict__ dst,
                              const float* __restrict__ ES, const float* __restrict__ ED,
                              const float* __restrict__ MX, float* __restrict__ EX,
                              float* __restrict__ DEN, int E, int ET) {
    int e = blockIdx.x * blockDim.x + threadIdx.x;
    if (e >= ET) return;
    int s = (e < E) ? src[e] : (e - E);
    int d = (e < E) ? dst[e] : (e - E);
#pragma unroll
    for (int h = 0; h < HEADS; ++h) {
        float v  = lrelu(ES[s * HEADS + h] + ED[d * HEADS + h]);
        float ex = expf(v - MX[d * HEADS + h]);
        EX[(size_t)e * HEADS + h] = ex;
        atomicAdd(&DEN[d * HEADS + h], ex);
    }
}

// ---- weighted aggregation: AGG[dst] += alpha * H[src] ----------------------
// one block (128 thr) per edge; thread c covers channel c of all 4 heads
__global__ void k_edge_agg(const int* __restrict__ src, const int* __restrict__ dst,
                           const float* __restrict__ EX, const float* __restrict__ DEN,
                           const float* __restrict__ H, float* __restrict__ AGG,
                           int E) {
    int e = blockIdx.x, c = threadIdx.x;
    int s = (e < E) ? src[e] : (e - E);
    int d = (e < E) ? dst[e] : (e - E);
#pragma unroll
    for (int h = 0; h < HEADS; ++h) {
        float alpha = EX[(size_t)e * HEADS + h] / DEN[d * HEADS + h];
        atomicAdd(&AGG[(size_t)d * NOUT + h * HID + c],
                  alpha * H[(size_t)s * NOUT + h * HID + c]);
    }
}

// ---- head mean + bias + relu; emit fp32 (pool) and bf16 (next GEMM A) ------
__global__ void k_finalize(const float* __restrict__ AGG, const float* __restrict__ b,
                           float* __restrict__ XN, unsigned short* __restrict__ A) {
    int n = blockIdx.x, c = threadIdx.x;
    const float* a = AGG + (size_t)n * NOUT;
    float v = 0.25f * (a[c] + a[c + 128] + a[c + 256] + a[c + 384]) + b[c];
    v = v > 0.0f ? v : 0.0f;
    XN[(size_t)n * HID + c] = v;
    A[(size_t)n * KP + c]   = f2bf(v);
}

// ---- pooling ---------------------------------------------------------------
__global__ void k_zero_pool(float* __restrict__ POOL, float* __restrict__ CNT, int G) {
    int i = blockIdx.x * blockDim.x + threadIdx.x;
    if (i < G * HID) POOL[i] = 0.0f;
    if (i < G)       CNT[i]  = 0.0f;
}

__global__ void k_pool(const float* __restrict__ XN, const int* __restrict__ batch,
                       float* __restrict__ POOL, float* __restrict__ CNT) {
    int n = blockIdx.x, c = threadIdx.x;
    int g = batch[n];
    atomicAdd(&POOL[g * HID + c], XN[(size_t)n * HID + c]);
    if (c == 0) atomicAdd(&CNT[g], 1.0f);
}

// ---- tiny MLP: [128 pooled | log1p(bs)] -> 128 -> 64 -> 2 ------------------
__global__ void k_mlp(const float* __restrict__ POOL, const float* __restrict__ CNT,
                      const float* __restrict__ bs,
                      const float* __restrict__ M1, const float* __restrict__ mb1,
                      const float* __restrict__ M2, const float* __restrict__ mb2,
                      const float* __restrict__ M3, const float* __restrict__ mb3,
                      float* __restrict__ out) {
    __shared__ float z[HID + 1], z1[HID], z2[HID / 2];
    int g = blockIdx.x, t = threadIdx.x;
    float cnt = CNT[g];
    cnt = cnt > 1.0f ? cnt : 1.0f;
    z[t] = POOL[g * HID + t] / cnt;
    if (t == 0) z[HID] = log1pf(bs[g]);
    __syncthreads();
    float a1 = mb1[t];
    for (int i = 0; i < HID + 1; ++i) a1 += z[i] * M1[(size_t)i * HID + t];
    z1[t] = a1 > 0.0f ? a1 : 0.0f;
    __syncthreads();
    if (t < HID / 2) {
        float a2 = mb2[t];
        for (int i = 0; i < HID; ++i) a2 += z1[i] * M2[(size_t)i * (HID / 2) + t];
        z2[t] = a2 > 0.0f ? a2 : 0.0f;
    }
    __syncthreads();
    if (t < 2) {
        float a3 = mb3[t];
        for (int i = 0; i < HID / 2; ++i) a3 += z2[i] * M3[(size_t)i * 2 + t];
        out[g * 2 + t] = a3;
    }
}

// ---------------------------------------------------------------------------
extern "C" void kernel_launch(void* const* d_in, const int* in_sizes, int n_in,
                              void* d_out, int out_size, void* d_ws, size_t ws_size,
                              hipStream_t stream) {
    const float* x     = (const float*)d_in[0];
    const int*   ei    = (const int*)d_in[1];
    const int*   batch = (const int*)d_in[2];
    const float* bs    = (const float*)d_in[3];
    const float* W[3]  = { (const float*)d_in[4],  (const float*)d_in[8],  (const float*)d_in[12] };
    const float* As[3] = { (const float*)d_in[5],  (const float*)d_in[9],  (const float*)d_in[13] };
    const float* Ad[3] = { (const float*)d_in[6],  (const float*)d_in[10], (const float*)d_in[14] };
    const float* Bb[3] = { (const float*)d_in[7],  (const float*)d_in[11], (const float*)d_in[15] };
    const float* M1  = (const float*)d_in[16];
    const float* mb1 = (const float*)d_in[17];
    const float* M2  = (const float*)d_in[18];
    const float* mb2 = (const float*)d_in[19];
    const float* M3  = (const float*)d_in[20];
    const float* mb3 = (const float*)d_in[21];

    const int N     = in_sizes[2];          // 20000 nodes (multiple of 16)
    const int E     = in_sizes[1] / 2;      // 320000 edges
    const int ET    = E + N;                // + self loops
    const int G     = in_sizes[3];          // 64 graphs
    const int inDim = in_sizes[4] / NOUT;   // 12

    const int* src = ei;
    const int* dst = ei + E;

    // ---- workspace carve (256B-aligned regions) ----------------------------
    size_t off = 0;
    auto carve = [&](size_t bytes) -> void* {
        void* r = (char*)d_ws + off;
        off += (bytes + 255) & ~(size_t)255;
        return r;
    };
    unsigned short* Abf = (unsigned short*)carve((size_t)N * KP * 2);   // 5.1 MB
    unsigned short* WT  = (unsigned short*)carve((size_t)NOUT * KP * 2);// 128 KB
    float* H   = (float*)carve((size_t)N * NOUT * 4);                   // 41 MB
    float* ES  = (float*)carve((size_t)N * HEADS * 4);
    float* ED  = (float*)carve((size_t)N * HEADS * 4);
    float* MX  = (float*)carve((size_t)N * HEADS * 4);
    float* DEN = (float*)carve((size_t)N * HEADS * 4);
    float* EX  = (float*)carve((size_t)ET * HEADS * 4);                 // 5.4 MB
    float* AGG = (float*)carve((size_t)N * NOUT * 4);                   // 41 MB
    float* XN  = (float*)carve((size_t)N * HID * 4);                    // 10 MB
    float* POOL = (float*)carve((size_t)G * HID * 4);
    float* CNT  = (float*)carve((size_t)G * 4);
    (void)ws_size; (void)n_in; (void)out_size;

    const long agg_total = (long)N * NOUT;
    const int  nh        = N * HEADS;

    // ---- layer 1 input cast -------------------------------------------------
    k_cast_x<<<N, HID, 0, stream>>>(x, Abf, inDim);

    for (int l = 0; l < 3; ++l) {
        const int Kw = (l == 0) ? inDim : HID;
        k_cast_wt<<<NOUT, KP, 0, stream>>>(W[l], WT, Kw);
        k_gemm_bf16<<<N / 16, 256, 0, stream>>>(Abf, WT, H);
        k_scores<<<N, 128, 0, stream>>>(H, As[l], Ad[l], ES, ED);
        k_init<<<(int)((agg_total + 255) / 256), 256, 0, stream>>>(AGG, MX, DEN, agg_total, nh);
        k_edge_max<<<(ET + 255) / 256, 256, 0, stream>>>(src, dst, ES, ED, MX, E, ET);
        k_edge_expsum<<<(ET + 255) / 256, 256, 0, stream>>>(src, dst, ES, ED, MX, EX, DEN, E, ET);
        k_edge_agg<<<ET, HID, 0, stream>>>(src, dst, EX, DEN, H, AGG, E);
        k_finalize<<<N, HID, 0, stream>>>(AGG, Bb[l], XN, Abf);
    }

    k_zero_pool<<<(G * (HID + 1) + 255) / 256, 256, 0, stream>>>(POOL, CNT, G);
    k_pool<<<N, HID, 0, stream>>>(XN, batch, POOL, CNT);
    k_mlp<<<G, HID, 0, stream>>>(POOL, CNT, bs, M1, mb1, M2, mb2, M3, mb3, (float*)d_out);
}
